// TransformerSelfAttentionPath_76484777607192
// MI455X (gfx1250) — matrix operand order repaired and checked by reading, not compiled, not run on hardware
//
#include <hip/hip_runtime.h>
#include <stdint.h>

// ---------------------------------------------------------------------------
// Problem constants (reference: B=2, S=512, H=768, NH=12, HD=64)
// ---------------------------------------------------------------------------
#define BB 2
#define SS 512
#define HH 768
#define NH 12
#define HD 64

typedef __attribute__((ext_vector_type(16))) __bf16 v16bf;
typedef __attribute__((ext_vector_type(2)))  __bf16 v2bf;
typedef __attribute__((ext_vector_type(8)))  float  v8f;

union __attribute__((aligned(16))) BF16Frag {
    uint32_t u[8];
    uint4    q[2];
    v16bf    v;
};

static __device__ inline uint16_t f2bf(float f) {
    uint32_t u = __builtin_bit_cast(uint32_t, f);
    uint32_t r = u + 0x7FFFu + ((u >> 16) & 1u);   // round-to-nearest-even
    return (uint16_t)(r >> 16);
}
static __device__ inline float bf2f(uint16_t h) {
    uint32_t u = ((uint32_t)h) << 16;
    return __builtin_bit_cast(float, u);
}
static __device__ inline uint32_t pack2(float lo, float hi) {
#if __has_builtin(__builtin_amdgcn_cvt_pk_bf16_f32)
    v2bf p = __builtin_amdgcn_cvt_pk_bf16_f32(lo, hi);
    return __builtin_bit_cast(uint32_t, p);
#else
    return (uint32_t)f2bf(lo) | ((uint32_t)f2bf(hi) << 16);
#endif
}

static __device__ inline v8f wmma_bf16(const BF16Frag& a, const BF16Frag& b, v8f c) {
    return __builtin_amdgcn_wmma_f32_16x16x32_bf16(
        /*neg_a=*/false, a.v, /*neg_b=*/false, b.v,
        /*c_mod=*/(short)0, c, /*reuse_a=*/false, /*reuse_b=*/false);
}

// ---------------------------------------------------------------------------
// Kernel 0: one-shot f32 -> bf16 conversion (pure bandwidth).
// Removes all conversion VALU from the GEMM hot loop and de-duplicates the
// work (previously every X tile was re-converted by 6 blocks, W by 8).
// ---------------------------------------------------------------------------
__global__ __launch_bounds__(256)
void f32_to_bf16_kernel(const float* __restrict__ in, uint16_t* __restrict__ outp, int n4) {
    int i = blockIdx.x * 256 + threadIdx.x;
    if (i < n4) {
        float4 v = ((const float4*)in)[i];
        uint2 o;
        o.x = pack2(v.x, v.y);
        o.y = pack2(v.z, v.w);
        ((uint2*)outp)[i] = o;
    }
}

// ---------------------------------------------------------------------------
// Kernel 1: Y = Xb @ Wb^T + bias  (bf16 inputs, M=1024, N=768, K=768)
//   mode 0: q  -> bf16 [B,NH,S,HD]
//   mode 1: k  -> bf16 [B,NH,S,HD]
//   mode 2: vt -> bf16 [B,NH,HD,S]   (transposed, for contiguous B-frags later)
//   mode 3: vf -> f32  [B,NH,S,HD]
// Block: 256 threads (8 waves), tile 128x128, K-step 32.
// Software-pipelined; LDS staged with straight uint4 copies (no conversion).
// ---------------------------------------------------------------------------
__global__ __launch_bounds__(256)
void proj_gemm_wmma(const uint16_t* __restrict__ Xb, const uint16_t* __restrict__ Wb,
                    const float* __restrict__ bias, void* __restrict__ out, int mode) {
    __shared__ uint32_t sX[128 * 16];   // 128 rows x 32 bf16 (packed dwords)
    __shared__ uint32_t sW[128 * 16];

    const int tid   = threadIdx.x;
    const int lane  = tid & 31;
    const int wid   = tid >> 5;
    const int wm    = wid & 3;          // 4 waves along M (32 rows each)
    const int wn    = wid >> 2;         // 2 waves along N (64 cols each)
    const int mbase = blockIdx.y * 128;
    const int nbase = blockIdx.x * 128;

    const uint32_t* X32 = (const uint32_t*)Xb;   // 384 dwords per 768-elem row
    const uint32_t* W32 = (const uint32_t*)Wb;

    v8f acc[2][4];
    v8f zero = {};
#pragma unroll
    for (int i = 0; i < 2; ++i)
#pragma unroll
        for (int j = 0; j < 4; ++j) acc[i][j] = zero;

    const int k0d = (lane < 16) ? 0 : 8;   // A-frag: K half (dwords)
    const int kbd = (lane < 16) ? 0 : 8;   // B-frag: K half (dwords)

    uint4 rX[2], rW[2];                 // staged 16B chunks (2 per operand tile)

    auto loadTile = [&](int k0) {       // k0 in elements; tile = 512 uint4
#pragma unroll
        for (int t = 0; t < 2; ++t) {
            int idx4 = tid * 2 + t;
            int row  = idx4 >> 2;       // 4 uint4 per row-slice of 32 bf16
            int c4   = idx4 & 3;
            rX[t] = *(const uint4*)&X32[(size_t)(mbase + row) * 384 + (k0 >> 1) + c4 * 4];
            rW[t] = *(const uint4*)&W32[(size_t)(nbase + row) * 384 + (k0 >> 1) + c4 * 4];
        }
    };
    auto storeTile = [&]() {
#pragma unroll
        for (int t = 0; t < 2; ++t) {
            int idx4 = tid * 2 + t;
            *(uint4*)&sX[idx4 * 4] = rX[t];
            *(uint4*)&sW[idx4 * 4] = rW[t];
        }
    };
    auto compute = [&]() {
        BF16Frag a[2];
#pragma unroll
        for (int amt = 0; amt < 2; ++amt) {
            const uint32_t* base = &sX[(wm * 32 + amt * 16 + (lane & 15)) * 16];
            a[amt].q[0] = *(const uint4*)(base + k0d);
            a[amt].q[1] = *(const uint4*)(base + k0d + 4);
        }
        BF16Frag bfr[4];
#pragma unroll
        for (int bnt = 0; bnt < 4; ++bnt) {
            const uint32_t* base = &sW[(wn * 64 + bnt * 16 + (lane & 15)) * 16];
            bfr[bnt].q[0] = *(const uint4*)(base + kbd);
            bfr[bnt].q[1] = *(const uint4*)(base + kbd + 4);
        }
#pragma unroll
        for (int amt = 0; amt < 2; ++amt)
#pragma unroll
            for (int bnt = 0; bnt < 4; ++bnt)
                acc[amt][bnt] = wmma_bf16(a[amt], bfr[bnt], acc[amt][bnt]);
    };

    // Pipeline: prologue stage, then {sync; prefetch next; compute; sync; store}.
    loadTile(0);
    storeTile();
    for (int k0 = 32; k0 < HH; k0 += 32) {
        __syncthreads();        // LDS tile visible to all waves
        loadTile(k0);           // prefetch next tile (overlaps with WMMAs below)
        compute();
        __syncthreads();        // everyone done reading LDS
        storeTile();
    }
    __syncthreads();
    compute();

    // Store C frags: VGPR j -> row j (lanes 0-15) / row j+8 (lanes 16-31), col = lane%16.
    const int rofs = (lane >= 16) ? 8 : 0;
    if (mode == 2) {            // vt transposed bf16 [B,NH,HD,S]
#pragma unroll
        for (int amt = 0; amt < 2; ++amt)
#pragma unroll
            for (int bnt = 0; bnt < 4; ++bnt) {
                int gn = nbase + wn * 64 + bnt * 16 + (lane & 15);
                float bv = bias[gn];
                int hh = gn >> 6, d = gn & 63;
#pragma unroll
                for (int j = 0; j < 8; ++j) {
                    int gm = mbase + wm * 32 + amt * 16 + j + rofs;
                    int bb = gm >> 9, s = gm & 511;
                    ((uint16_t*)out)[(((size_t)bb * NH + hh) * HD + d) * SS + s] =
                        f2bf(acc[amt][bnt][j] + bv);
                }
            }
    } else if (mode == 3) {     // vf f32 [B,NH,S,HD]
#pragma unroll
        for (int amt = 0; amt < 2; ++amt)
#pragma unroll
            for (int bnt = 0; bnt < 4; ++bnt) {
                int gn = nbase + wn * 64 + bnt * 16 + (lane & 15);
                float bv = bias[gn];
                int hh = gn >> 6, d = gn & 63;
#pragma unroll
                for (int j = 0; j < 8; ++j) {
                    int gm = mbase + wm * 32 + amt * 16 + j + rofs;
                    int bb = gm >> 9, s = gm & 511;
                    ((float*)out)[(((size_t)bb * NH + hh) * SS + s) * HD + d] =
                        acc[amt][bnt][j] + bv;
                }
            }
    } else {                    // q / k bf16 [B,NH,S,HD]
#pragma unroll
        for (int amt = 0; amt < 2; ++amt)
#pragma unroll
            for (int bnt = 0; bnt < 4; ++bnt) {
                int gn = nbase + wn * 64 + bnt * 16 + (lane & 15);
                float bv = bias[gn];
                int hh = gn >> 6, d = gn & 63;
#pragma unroll
                for (int j = 0; j < 8; ++j) {
                    int gm = mbase + wm * 32 + amt * 16 + j + rofs;
                    int bb = gm >> 9, s = gm & 511;
                    ((uint16_t*)out)[(((size_t)bb * NH + hh) * SS + s) * HD + d] =
                        f2bf(acc[amt][bnt][j] + bv);
                }
            }
    }
}

// ---------------------------------------------------------------------------
// Kernel 2: attention for one (b, h, 32-query tile). 2 waves x 16 rows.
//   scores = q k^T / 8 + mask -> LDS (32x512 f32, 64KB)
//   softmax in place, probs packed bf16 into low half of each row,
//   rowsum stashed at dword 256 of each row,
//   ctx = probs @ vt^T + rowsum * vf  -> out f32 [B,S,H]
// ---------------------------------------------------------------------------
__global__ __launch_bounds__(64)
void attn_wmma(const uint16_t* __restrict__ qb, const uint16_t* __restrict__ kb,
               const uint16_t* __restrict__ vtT, const float* __restrict__ vf,
               const float* __restrict__ mask, float* __restrict__ out) {
    __shared__ float sS[32 * 512];              // exactly 64 KB
    uint32_t* sU = (uint32_t*)sS;

    const int tid  = threadIdx.x;
    const int lane = tid & 31;
    const int wid  = tid >> 5;                  // 0..1
    const int qt   = blockIdx.x;                // 0..15
    const int h    = blockIdx.y;                // 0..11
    const int b    = blockIdx.z;                // 0..1

    const size_t headBase = (size_t)(b * NH + h) * SS;   // row offset in [B,NH,S,*]
    const int rofs = (lane >= 16) ? 8 : 0;
    const int k0d  = (lane < 16) ? 0 : 8;       // A-frag K half (dwords)
    const int kbd  = (lane < 16) ? 0 : 8;       // B-frag K half (dwords)

    // ---- Phase 1: scores = q k^T * 0.125 + mask --------------------------
    const uint32_t* q32 = (const uint32_t*)qb;  // 32 dwords per 64-elem row
    const uint32_t* k32 = (const uint32_t*)kb;

    BF16Frag aq[2];
    {
        int qrow = qt * 32 + wid * 16 + (lane & 15);
        const uint32_t* qr = q32 + (headBase + qrow) * 32;
#pragma unroll
        for (int kk = 0; kk < 2; ++kk) {        // head-dim halves 0..31 / 32..63
            aq[kk].q[0] = *(const uint4*)(qr + kk * 16 + k0d);
            aq[kk].q[1] = *(const uint4*)(qr + kk * 16 + k0d + 4);
        }
    }

    for (int kt = 0; kt < 32; ++kt) {           // 32 key tiles of 16
        int krow = kt * 16 + (lane & 15);
        const uint32_t* kr = k32 + (headBase + krow) * 32;
        BF16Frag bk[2];
#pragma unroll
        for (int kk = 0; kk < 2; ++kk) {
            bk[kk].q[0] = *(const uint4*)(kr + kk * 16 + kbd);
            bk[kk].q[1] = *(const uint4*)(kr + kk * 16 + kbd + 4);
        }
        v8f c = {};
        c = wmma_bf16(aq[0], bk[0], c);
        c = wmma_bf16(aq[1], bk[1], c);

        float mval = mask[(size_t)b * SS + kt * 16 + (lane & 15)];
#pragma unroll
        for (int j = 0; j < 8; ++j) {
            int rl = wid * 16 + j + rofs;
            sS[rl * 512 + kt * 16 + (lane & 15)] = c[j] * 0.125f + mval;
        }
    }
    __syncthreads();

    // ---- Phase 2: softmax (one thread per row), pack bf16 in place -------
    if (tid < 32) {
        float* row = &sS[tid * 512];
        float m = -3.402823466e+38f;
        for (int i = 0; i < 512; ++i) m = fmaxf(m, row[i]);
        float s = 0.f;
        for (int i = 0; i < 512; ++i) { float e = __expf(row[i] - m); row[i] = e; s += e; }
        float inv = 1.0f / s;
        uint32_t* pr = (uint32_t*)row;
        float rs = 0.f;
        for (int i = 0; i < 256; ++i) {         // write dword i after reading 2i,2i+1
            uint16_t p0 = f2bf(row[2 * i] * inv);
            uint16_t p1 = f2bf(row[2 * i + 1] * inv);
            pr[i] = (uint32_t)p0 | ((uint32_t)p1 << 16);
            rs += bf2f(p0) + bf2f(p1);
        }
        row[256] = rs;                          // stash rowsum in freed high half
    }
    __syncthreads();

    // ---- Phase 3: ctx = probs @ vt^T  (K over 512 keys) ------------------
    const uint32_t* vt32 = (const uint32_t*)vtT;  // [B,NH,HD,S] -> 256 dwords per d-row
    const size_t vtHead = (size_t)(b * NH + h) * HD;

    v8f cc[4];
    v8f zero = {};
#pragma unroll
    for (int nt = 0; nt < 4; ++nt) cc[nt] = zero;

    const int prow = (wid * 16 + (lane & 15)) * 512;   // probs row base (dwords)
    for (int kt = 0; kt < 16; ++kt) {           // K-steps of 32 keys
        BF16Frag ap;
        ap.q[0] = *(const uint4*)(sU + prow + kt * 16 + k0d);
        ap.q[1] = *(const uint4*)(sU + prow + kt * 16 + k0d + 4);
#pragma unroll
        for (int nt = 0; nt < 4; ++nt) {
            int d = nt * 16 + (lane & 15);
            const uint32_t* vr = vt32 + (vtHead + d) * 256;
            BF16Frag bv;
            bv.q[0] = *(const uint4*)(vr + kt * 16 + kbd);
            bv.q[1] = *(const uint4*)(vr + kt * 16 + kbd + 4);
            cc[nt] = wmma_bf16(ap, bv, cc[nt]);
        }
    }

    // ---- Epilogue: + rowsum * vf, store [B,S,H] f32 ----------------------
#pragma unroll
    for (int nt = 0; nt < 4; ++nt) {
        int d = nt * 16 + (lane & 15);
#pragma unroll
        for (int j = 0; j < 8; ++j) {
            int rl = wid * 16 + j + rofs;
            int qg = qt * 32 + rl;
            float rs = sS[rl * 512 + 256];
            float val = cc[nt][j] + rs * vf[(headBase + qg) * HD + d];
            out[((size_t)b * SS + qg) * HH + h * HD + d] = val;
        }
    }
}

// ---------------------------------------------------------------------------
// Host launcher
// ---------------------------------------------------------------------------
extern "C" void kernel_launch(void* const* d_in, const int* in_sizes, int n_in,
                              void* d_out, int out_size, void* d_ws, size_t ws_size,
                              hipStream_t stream) {
    (void)in_sizes; (void)n_in; (void)out_size; (void)ws_size;

    const float* from = (const float*)d_in[0];
    const float* to   = (const float*)d_in[1];
    const float* mask = (const float*)d_in[2];
    const float* Wq   = (const float*)d_in[3];
    const float* bq   = (const float*)d_in[4];
    const float* Wk   = (const float*)d_in[5];
    const float* bk   = (const float*)d_in[6];
    const float* Wvf  = (const float*)d_in[7];
    const float* bvf  = (const float*)d_in[8];
    const float* Wvt  = (const float*)d_in[9];
    const float* bvt  = (const float*)d_in[10];

    const size_t nAct = (size_t)BB * SS * HH;          // 786432 (from / to)
    const size_t nW   = (size_t)HH * HH;               // 589824 (each weight)
    const size_t nElem = (size_t)BB * NH * SS * HD;    // 786432 (per head tensor)

    // Workspace layout (bf16 unless noted):
    uint16_t* qb    = (uint16_t*)d_ws;                 // [B,NH,S,HD]
    uint16_t* kbuf  = qb   + nElem;                    // [B,NH,S,HD]
    uint16_t* vtT   = kbuf + nElem;                    // [B,NH,HD,S]
    float*    vf    = (float*)(vtT + nElem);           // [B,NH,S,HD] f32
    uint16_t* fromB = (uint16_t*)(vf + nElem);         // [B,S,H]
    uint16_t* toB   = fromB + nAct;                    // [B,S,H]
    uint16_t* WqB   = toB   + nAct;                    // [H,H]
    uint16_t* WkB   = WqB   + nW;
    uint16_t* WvfB  = WkB   + nW;
    uint16_t* WvtB  = WvfB  + nW;

    // Stage 0: one-shot f32 -> bf16 conversions (pure bandwidth).
    {
        int a4 = (int)(nAct / 4), w4 = (int)(nW / 4);
        dim3 ga((a4 + 255) / 256), gw((w4 + 255) / 256);
        f32_to_bf16_kernel<<<ga, 256, 0, stream>>>(from, fromB, a4);
        f32_to_bf16_kernel<<<ga, 256, 0, stream>>>(to,   toB,   a4);
        f32_to_bf16_kernel<<<gw, 256, 0, stream>>>(Wq,  WqB,  w4);
        f32_to_bf16_kernel<<<gw, 256, 0, stream>>>(Wk,  WkB,  w4);
        f32_to_bf16_kernel<<<gw, 256, 0, stream>>>(Wvf, WvfB, w4);
        f32_to_bf16_kernel<<<gw, 256, 0, stream>>>(Wvt, WvtB, w4);
    }

    // Stage 1: four WMMA projections.
    dim3 gridP(HH / 128, (BB * SS) / 128);             // 6 x 8
    proj_gemm_wmma<<<gridP, 256, 0, stream>>>(fromB, WqB,  bq,  (void*)qb,   0);
    proj_gemm_wmma<<<gridP, 256, 0, stream>>>(toB,   WkB,  bk,  (void*)kbuf, 1);
    proj_gemm_wmma<<<gridP, 256, 0, stream>>>(toB,   WvtB, bvt, (void*)vtT,  2);
    proj_gemm_wmma<<<gridP, 256, 0, stream>>>(fromB, WvfB, bvf, (void*)vf,   3);

    // Stage 2: fused attention.
    dim3 gridA(SS / 32, NH, BB);                       // 16 x 12 x 2
    attn_wmma<<<gridA, 64, 0, stream>>>(qb, kbuf, vtT, vf, mask, (float*)d_out);
}
